// SAE_35527969472628
// MI455X (gfx1250) — matrix operand order repaired and checked by reading, not compiled, not run on hardware
//
#include <hip/hip_runtime.h>

// ---------------- types / constants ----------------
typedef _Float16 half_t;
typedef __attribute__((ext_vector_type(16))) _Float16 v16h;
typedef __attribute__((ext_vector_type(8)))  _Float16 v8hh;
typedef __attribute__((ext_vector_type(8)))  float    v8f;
typedef __attribute__((ext_vector_type(4)))  unsigned int u32x4;
typedef __attribute__((ext_vector_type(4)))  int i32x4;
typedef __attribute__((ext_vector_type(8)))  int i32x8;

#define TT 50
#define BB 16
#define SZD 64
#define C1C 16
#define C2C 16
#define KW 5
#define S1D 60
#define S2D 56
#define SP1 (S1D*S1D)       // 3600
#define SP2 (S2D*S2D)       // 3136
#define NCONV2 (C2C*SP2)    // 50176
#define NREC 256
#define KPACK 512           // conv GEMM K: 16 channels x 32 (25 taps + 7 zero pad)
#define KS 8                // split-K factor for the ff_in GEMM
#define KSLICE (NCONV2/KS)  // 6272 (multiple of 32)
#define STG 128             // halfs per row per TDM stage (16 rows x 128 h = 4KB)
#define NSTG (KSLICE/STG)   // 49 stages
#define BETA 0.90483741803595952f
#define THRESH 1.0f

#if defined(__has_builtin)
#  if __has_builtin(__builtin_amdgcn_tensor_load_to_lds) && \
      __has_builtin(__builtin_amdgcn_s_wait_tensorcnt)
#    define HAVE_TDM 1
#  else
#    define HAVE_TDM 0
#  endif
#else
#  define HAVE_TDM 0
#endif

// D = A(16x32 f16) * B(32x16 f16) + C(16x16 f32), wave32 WMMA
__device__ __forceinline__ v8f wmma16(v16h a, v16h b, v8f c) {
  return __builtin_amdgcn_wmma_f32_16x16x32_f16(false, a, false, b, (short)0, c,
                                                false, false);
}

// One 16x16x32 MAC step. ap -> row-major A at [m][kb+16*hs] (one 32B run),
// bp -> transposed-B (N x K row-major) at [n][kb+16*hs] (one 32B run).
__device__ __forceinline__ void wmma_step(const half_t* __restrict__ ap,
                                          const half_t* __restrict__ bp, v8f& acc) {
  v16h a, b;
  a = *(const v16h*)(ap);                     // K = kb+16hs   .. +15
  b = *(const v16h*)(bp);                     // K = kb+16hs   .. +15
  acc = wmma16(a, b, acc);
}

#if HAVE_TDM
// TDM: DMA a 16-row x `cols`-halfs 2D tile (row stride `rowstride_elts` halfs)
// from global into LDS at byte offset lds_off. D# per ISA ch.8: count=1,
// type=2, data_size=2B; descriptor words forced uniform -> SGPR groups.
__device__ __forceinline__ void tdm_load_tile16(unsigned lds_off, const half_t* g,
                                                int cols, int rowstride_elts) {
  unsigned long long ga = (unsigned long long)(void*)g;
  u32x4 g0;
  g0[0] = (unsigned)__builtin_amdgcn_readfirstlane(1);              // count=1 valid D#
  g0[1] = (unsigned)__builtin_amdgcn_readfirstlane((int)lds_off);   // lds_addr (bytes)
  g0[2] = (unsigned)__builtin_amdgcn_readfirstlane((int)(unsigned)ga);
  g0[3] = (unsigned)__builtin_amdgcn_readfirstlane(
              (int)(((unsigned)(ga >> 32) & 0x1FFFFFFu) | 0x80000000u)); // addr[56:32]|type=2
  i32x8 g1;
  g1[0] = __builtin_amdgcn_readfirstlane(1 << 16);                  // data_size = 2 bytes
  g1[1] = __builtin_amdgcn_readfirstlane(cols << 16);               // tensor_dim0[15:0]
  g1[2] = __builtin_amdgcn_readfirstlane(16 << 16);                 // tensor_dim1 = 16
  g1[3] = __builtin_amdgcn_readfirstlane(cols << 16);               // tile_dim0
  g1[4] = __builtin_amdgcn_readfirstlane(16);                       // tile_dim1 = 16
  g1[5] = __builtin_amdgcn_readfirstlane(rowstride_elts);           // tensor_dim0_stride
  g1[6] = 0;
  g1[7] = 0;
  i32x4 gz = {0, 0, 0, 0};
#if __clang_major__ >= 23
  i32x8 gz8 = {0, 0, 0, 0, 0, 0, 0, 0};
  __builtin_amdgcn_tensor_load_to_lds(g0, g1, gz, gz, gz8, 0);
#else
  __builtin_amdgcn_tensor_load_to_lds(g0, g1, gz, gz, 0);
#endif
}
#endif

// ---------------- init / prep kernels ----------------
__global__ void k_zero_f(float* p, int n) {
  int i = blockIdx.x * blockDim.x + threadIdx.x;
  if (i < n) p[i] = 0.0f;
}
__global__ void k_zero_h(half_t* p, int n) {
  int i = blockIdx.x * blockDim.x + threadIdx.x;
  if (i < n) p[i] = (half_t)0.0f;
}
// plain f32 -> f16 convert; PyTorch [out][in] weight layout IS the transposed-B
// layout the WMMA B-fragment wants, so no transpose kernels are needed.
__global__ void k_f2h(const float* __restrict__ s, half_t* __restrict__ d, int n) {
  int i = blockIdx.x * blockDim.x + threadIdx.x;
  if (i < n) d[i] = (half_t)s[i];
}
// conv2_w (C2,C1,5,5) -> A-matrix [m][c1*32 + tap], tap = ky*5+kx, zero padded
__global__ void k_prep_w2p(const float* __restrict__ w, half_t* __restrict__ d) {
  int i = blockIdx.x * blockDim.x + threadIdx.x;
  if (i >= C2C * KPACK) return;
  int m = i >> 9, r = i & 511, c1 = r >> 5, tap = r & 31;
  d[i] = (tap < 25) ? (half_t)w[(m * C1C + c1) * 25 + tap] : (half_t)0.0f;
}
// deconv2_w [Cin=C2][Cout=C1][5][5] -> flipped A-matrix [o][c2*32 + tap]
__global__ void k_prep_wd2p(const float* __restrict__ w, half_t* __restrict__ d) {
  int i = blockIdx.x * blockDim.x + threadIdx.x;
  if (i >= C1C * KPACK) return;
  int m = i >> 9, r = i & 511, c2 = r >> 5, tap = r & 31;
  if (tap < 25) {
    int ky = tap / 5, kx = tap % 5;
    d[i] = (half_t)w[((c2 * C1C + m) * KW + (KW - 1 - ky)) * KW + (KW - 1 - kx)];
  } else {
    d[i] = (half_t)0.0f;
  }
}
// recon_w [Cin=C1][1][5][5] -> flipped float taps [c1*25 + ky*5 + kx]
__global__ void k_prep_wrc(const float* __restrict__ w, float* __restrict__ d) {
  int i = blockIdx.x * blockDim.x + threadIdx.x;
  if (i >= C1C * KW * KW) return;
  int c = i / 25, rr = i % 25, ky = rr / 5, kx = rr % 5;
  d[i] = w[(c * KW + (KW - 1 - ky)) * KW + (KW - 1 - kx)];
}

// ---------------- per-step kernels ----------------
// conv1 (1->16ch, 5x5 VALID) + LIF ; 46 MFLOP/step -> plain VALU
__global__ void k_conv1_lif(const float* __restrict__ xt, const float* __restrict__ w,
                            const float* __restrict__ bias, const float* __restrict__ osc,
                            int t, float* __restrict__ mem, half_t* __restrict__ spk1) {
  int i = blockIdx.x * blockDim.x + threadIdx.x;   // BB*C1C*SP1 = 921600
  if (i >= BB * C1C * SP1) return;
  int b = i / (C1C * SP1);
  int r = i % (C1C * SP1);
  int c = r / SP1;
  int sp = r % SP1;
  int y = sp / S1D, x = sp % S1D;
  const float* xb = xt + (size_t)b * SZD * SZD;
  const float* wc = w + c * KW * KW;
  float acc = bias[c] + osc[t];
#pragma unroll
  for (int ky = 0; ky < KW; ++ky)
#pragma unroll
    for (int kx = 0; kx < KW; ++kx)
      acc += wc[ky * KW + kx] * xb[(y + ky) * SZD + (x + kx)];
  float mv = mem[i];
  float rst = mv > THRESH ? 1.0f : 0.0f;
  float mn = BETA * mv + acc - rst;
  mem[i] = mn;
  spk1[i] = (half_t)(mn > THRESH ? 1.0f : 0.0f);
}

// im2col pack for conv2: bp[n*512 + c1*32 + tap] = spk1[b][c1][y+ky][x+kx]
__global__ void k_pack_conv2(const half_t* __restrict__ spk1, half_t* __restrict__ bp) {
  int i = blockIdx.x * blockDim.x + threadIdx.x;   // C1C * BB*SP2 = 802816
  if (i >= C1C * BB * SP2) return;
  int c1 = i / (BB * SP2);
  int n = i % (BB * SP2);
  int b = n / SP2, sp = n % SP2, y = sp / S2D, x = sp % S2D;
  const half_t* src = spk1 + ((size_t)(b * C1C + c1) * S1D + y) * S1D + x;
  half_t* dst = bp + (size_t)n * KPACK + c1 * 32;
#pragma unroll
  for (int ky = 0; ky < KW; ++ky)
#pragma unroll
    for (int kx = 0; kx < KW; ++kx)
      dst[ky * 5 + kx] = src[ky * S1D + kx];
#pragma unroll
  for (int z = 25; z < 32; ++z) dst[z] = (half_t)0.0f;
}

// im2col pack for deconv2 from the zero-padded (B,C2,64,64) buffer
__global__ void k_pack_deconv2(const half_t* __restrict__ cpad, half_t* __restrict__ bp) {
  int i = blockIdx.x * blockDim.x + threadIdx.x;   // C2C * BB*SP1 = 921600
  if (i >= C2C * BB * SP1) return;
  int c2 = i / (BB * SP1);
  int n = i % (BB * SP1);
  int b = n / SP1, sp = n % SP1, Y = sp / S1D, X = sp % S1D;
  const half_t* src = cpad + ((size_t)(b * C2C + c2) * SZD + Y) * SZD + X;
  half_t* dst = bp + (size_t)n * KPACK + c2 * 32;
#pragma unroll
  for (int ky = 0; ky < KW; ++ky)
#pragma unroll
    for (int kx = 0; kx < KW; ++kx)
      dst[ky * 5 + kx] = src[ky * SZD + kx];
#pragma unroll
  for (int z = 25; z < 32; ++z) dst[z] = (half_t)0.0f;
}

// conv2 / deconv2 GEMM core: M=16 chans, K=512 packed, N tiles of 16 ; WMMA + LIF
// SPD is compile-time so the epilogue decode is magic-multiply, not runtime div.
template <int SPD>
__global__ void k_conv_gemm_lif(const half_t* __restrict__ wA,   // [16][512]
                                const half_t* __restrict__ bpk,  // [n][512]
                                const float* __restrict__ bias,
                                const float* __restrict__ osc, int t,
                                float* __restrict__ mem, half_t* __restrict__ spk) {
  const int lane = threadIdx.x & 31;
  const int wave = threadIdx.x >> 5;
  const int tile = blockIdx.x * (blockDim.x >> 5) + wave;
  const int lm = lane & 15, hs = lane >> 4;
  const int n = tile * 16 + lm;
  const int b = n / SPD, sp = n % SPD;
  const half_t* ap = wA + lm * KPACK + 16 * hs;
  const half_t* bp = bpk + (size_t)n * KPACK + 16 * hs;
  v8f acc = {0.f, 0.f, 0.f, 0.f, 0.f, 0.f, 0.f, 0.f};
#pragma unroll 4
  for (int kb = 0; kb < KPACK; kb += 32) wmma_step(ap + kb, bp + kb, acc);
  float o = osc[t];
#pragma unroll
  for (int r = 0; r < 8; ++r) {
    int m = r + 8 * hs;                       // output channel
    int gi = (b * 16 + m) * SPD + sp;
    float mv = mem[gi];
    float rst = mv > THRESH ? 1.0f : 0.0f;
    float mn = BETA * mv + acc[r] + bias[m] + o - rst;
    mem[gi] = mn;
    spk[gi] = (half_t)(mn > THRESH ? 1.0f : 0.0f);
  }
}

// ff_in (16x50176x256) split-K x8 + ff_rec (16x256x256) ; partials, no atomics.
// TDM path: each wave DMAs its private 16x128 B-panel stages into LDS
// (double-buffered, TENSORcnt-synced) while A comes straight from L2.
__global__ void __launch_bounds__(256)
k_ff_gemm(const half_t* __restrict__ spk2, const half_t* __restrict__ wfi,
          const half_t* __restrict__ srp, const half_t* __restrict__ wfr,
          float* __restrict__ partial) {
  const int lane = threadIdx.x & 31;
  const int wave = threadIdx.x >> 5;
  const int gid = blockIdx.x * (blockDim.x >> 5) + wave;  // 128 waves
  const int tile = gid & 15;    // N tile (16 columns)
  const int ks = gid >> 4;      // K slice
  const int lm = lane & 15, hs = lane >> 4;
  const int n = tile * 16 + lm;
  const half_t* ap = spk2 + (size_t)lm * NCONV2 + ks * KSLICE + 16 * hs;
  v8f acc = {0.f, 0.f, 0.f, 0.f, 0.f, 0.f, 0.f, 0.f};
#if HAVE_TDM
  __shared__ __align__(32) half_t lds[8][2][16 * STG];    // 64 KB / workgroup
  const half_t* gB = wfi + (size_t)(tile * 16) * NCONV2 + ks * KSLICE;
  unsigned lbase = (unsigned)(unsigned long long)(void*)&lds[wave][0][0];
  tdm_load_tile16(lbase, gB, STG, NCONV2);                // prologue stage 0
  for (int s = 0; s < NSTG; ++s) {
    if (s + 1 < NSTG) {
      tdm_load_tile16(lbase + (unsigned)(((s + 1) & 1) * (16 * STG * 2)),
                      gB + (size_t)(s + 1) * STG, STG, NCONV2);
      __builtin_amdgcn_s_wait_tensorcnt(1);               // stage s landed
    } else {
      __builtin_amdgcn_s_wait_tensorcnt(0);
    }
    const half_t* bp = &lds[wave][s & 1][lm * STG + 16 * hs];
    const half_t* as = ap + (size_t)s * STG;
#pragma unroll
    for (int kc = 0; kc < STG; kc += 32) wmma_step(as + kc, bp + kc, acc);
  }
#else
  const half_t* bp = wfi + (size_t)n * NCONV2 + ks * KSLICE + 16 * hs;
#pragma unroll 4
  for (int kb = 0; kb < KSLICE; kb += 32) {
    __builtin_prefetch(bp + kb + 512, 0, 1);
    wmma_step(ap + kb, bp + kb, acc);
  }
#endif
  if (ks == 0) {  // wave-uniform branch: EXEC stays all-ones for WMMA
    const half_t* ap2 = srp + lm * NREC + 16 * hs;
    const half_t* bp2 = wfr + (size_t)n * NREC + 16 * hs;
#pragma unroll
    for (int kb = 0; kb < NREC; kb += 32) wmma_step(ap2 + kb, bp2 + kb, acc);
  }
#pragma unroll
  for (int r = 0; r < 8; ++r)
    partial[(size_t)ks * BB * NREC + (r + 8 * hs) * NREC + n] = acc[r];
}

// deterministic split-K reduction + zero-reset LIF
__global__ void k_ff_lif_reduce(const float* __restrict__ partial,
                                const float* __restrict__ bi, const float* __restrict__ br,
                                const float* __restrict__ osc, int t,
                                float* __restrict__ mem, half_t* __restrict__ srn,
                                float* __restrict__ out_rec) {
  int i = blockIdx.x * blockDim.x + threadIdx.x;   // BB*NREC = 4096
  if (i >= BB * NREC) return;
  float s = 0.0f;
#pragma unroll
  for (int ks = 0; ks < KS; ++ks) s += partial[(size_t)ks * BB * NREC + i];
  int n = i & (NREC - 1);
  float mv = mem[i];
  float rst = mv > THRESH ? 1.0f : 0.0f;
  float base = BETA * mv + s + bi[n] + br[n] + osc[t];
  float mn = base * (1.0f - rst);                  // zero reset
  float spk = mn > THRESH ? 1.0f : 0.0f;
  mem[i] = mn;
  srn[i] = (half_t)spk;
  out_rec[i] = spk;
}

// ff_out (16x256x50176) -> zero-padded f16 (B,C2,64,64) ; WMMA
__global__ void k_ffout_pad(const half_t* __restrict__ src, const half_t* __restrict__ wfo,
                            const float* __restrict__ bias, half_t* __restrict__ cpad) {
  const int lane = threadIdx.x & 31;
  const int wave = threadIdx.x >> 5;
  const int tile = blockIdx.x * (blockDim.x >> 5) + wave;   // 3136 tiles
  const int lm = lane & 15, hs = lane >> 4;
  const int j = tile * 16 + lm;
  const int c = j / SP2, sp = j % SP2, y = sp / S2D, x = sp % S2D;
  const half_t* ap = src + lm * NREC + 16 * hs;
  const half_t* bp = wfo + (size_t)j * NREC + 16 * hs;
  v8f acc = {0.f, 0.f, 0.f, 0.f, 0.f, 0.f, 0.f, 0.f};
#pragma unroll
  for (int kb = 0; kb < NREC; kb += 32) wmma_step(ap + kb, bp + kb, acc);
  float bj = bias[j];
#pragma unroll
  for (int r = 0; r < 8; ++r) {
    int b = r + 8 * hs;                                     // batch
    cpad[((b * C2C + c) * SZD + (y + 4)) * SZD + (x + 4)] = (half_t)(acc[r] + bj);
  }
}

// recon deconv (16ch -> 1ch, pad 4) + LIF ; 52 MFLOP/step -> VALU
__global__ void k_recon_lif(const half_t* __restrict__ spkd2, const float* __restrict__ wrc,
                            const float* __restrict__ bias, const float* __restrict__ osc,
                            int t, float* __restrict__ mem, float* __restrict__ out_rc) {
  int i = blockIdx.x * blockDim.x + threadIdx.x;            // BB*64*64 = 65536
  if (i >= BB * SZD * SZD) return;
  int b = i / (SZD * SZD), sp = i % (SZD * SZD), Y = sp / SZD, X = sp % SZD;
  float acc = bias[0] + osc[t];
  for (int c = 0; c < C1C; ++c) {
#pragma unroll
    for (int ky = 0; ky < KW; ++ky) {
      int iy = Y + ky - 4;
      if (iy < 0 || iy >= S1D) continue;
#pragma unroll
      for (int kx = 0; kx < KW; ++kx) {
        int ix = X + kx - 4;
        if (ix < 0 || ix >= S1D) continue;
        acc += wrc[c * 25 + ky * 5 + kx] *
               (float)spkd2[((b * C1C + c) * S1D + iy) * S1D + ix];
      }
    }
  }
  float mv = mem[i];
  float rst = mv > THRESH ? 1.0f : 0.0f;
  float mn = BETA * mv + acc - rst;
  mem[i] = mn;
  out_rc[i] = (mn > THRESH) ? 1.0f : 0.0f;
}

// ---------------- host launcher ----------------
extern "C" void kernel_launch(void* const* d_in, const int* in_sizes, int n_in,
                              void* d_out, int out_size, void* d_ws, size_t ws_size,
                              hipStream_t stream) {
  (void)in_sizes; (void)n_in; (void)out_size; (void)ws_size;
  const float* xin      = (const float*)d_in[0];
  const float* osc      = (const float*)d_in[1];
  const float* conv1_w  = (const float*)d_in[2];
  const float* conv1_b  = (const float*)d_in[3];
  const float* conv2_w  = (const float*)d_in[4];
  const float* conv2_b  = (const float*)d_in[5];
  const float* ff_in_w  = (const float*)d_in[6];
  const float* ff_in_b  = (const float*)d_in[7];
  const float* ff_rec_w = (const float*)d_in[8];
  const float* ff_rec_b = (const float*)d_in[9];
  const float* ff_out_w = (const float*)d_in[10];
  const float* ff_out_b = (const float*)d_in[11];
  const float* deconv2_w= (const float*)d_in[12];
  const float* deconv2_b= (const float*)d_in[13];
  const float* recon_w  = (const float*)d_in[14];
  const float* recon_b  = (const float*)d_in[15];
  float* out = (float*)d_out;

  // workspace carve-up
  char* p = (char*)d_ws;
  auto alloc = [&](size_t bytes) -> char* {
    char* r = p;
    p += (bytes + 255) & ~(size_t)255;
    return r;
  };
  float*  mem_c1 = (float*)alloc(sizeof(float) * BB * C1C * SP1);
  float*  mem_c2 = (float*)alloc(sizeof(float) * BB * NCONV2);
  float*  mem_rec= (float*)alloc(sizeof(float) * BB * NREC);
  float*  mem_d2 = (float*)alloc(sizeof(float) * BB * C1C * SP1);
  float*  mem_rc = (float*)alloc(sizeof(float) * BB * SZD * SZD);
  float*  partial= (float*)alloc(sizeof(float) * KS * BB * NREC);
  half_t* srec0  = (half_t*)alloc(sizeof(half_t) * BB * NREC);
  half_t* srec1  = (half_t*)alloc(sizeof(half_t) * BB * NREC);
  half_t* spk1   = (half_t*)alloc(sizeof(half_t) * BB * C1C * SP1);
  half_t* spk2   = (half_t*)alloc(sizeof(half_t) * BB * NCONV2);
  half_t* cpad   = (half_t*)alloc(sizeof(half_t) * BB * C2C * SZD * SZD);
  half_t* spkd2  = (half_t*)alloc(sizeof(half_t) * BB * C1C * SP1);
  half_t* w2p    = (half_t*)alloc(sizeof(half_t) * C2C * KPACK);
  half_t* wd2p   = (half_t*)alloc(sizeof(half_t) * C1C * KPACK);
  float*  wrcf   = (float*)alloc(sizeof(float) * C1C * KW * KW);
  half_t* wfih   = (half_t*)alloc(sizeof(half_t) * (size_t)NREC * NCONV2);   // as-is layout
  half_t* wfrh   = (half_t*)alloc(sizeof(half_t) * NREC * NREC);
  half_t* wfoh   = (half_t*)alloc(sizeof(half_t) * (size_t)NCONV2 * NREC);   // as-is layout
  half_t* bpack  = (half_t*)alloc(sizeof(half_t) * (size_t)BB * SP1 * KPACK); // shared im2col
  half_t* srec[2] = {srec0, srec1};

  const dim3 blk(256);
  // zero membrane state + recurrent spikes + padded buffer borders
  k_zero_f<<<(BB * C1C * SP1 + 255) / 256, blk, 0, stream>>>(mem_c1, BB * C1C * SP1);
  k_zero_f<<<(BB * NCONV2 + 255) / 256, blk, 0, stream>>>(mem_c2, BB * NCONV2);
  k_zero_f<<<(BB * NREC + 255) / 256, blk, 0, stream>>>(mem_rec, BB * NREC);
  k_zero_f<<<(BB * C1C * SP1 + 255) / 256, blk, 0, stream>>>(mem_d2, BB * C1C * SP1);
  k_zero_f<<<(BB * SZD * SZD + 255) / 256, blk, 0, stream>>>(mem_rc, BB * SZD * SZD);
  k_zero_h<<<(BB * NREC + 255) / 256, blk, 0, stream>>>(srec0, BB * NREC);
  k_zero_h<<<(BB * NREC + 255) / 256, blk, 0, stream>>>(srec1, BB * NREC);
  k_zero_h<<<(BB * C2C * SZD * SZD + 255) / 256, blk, 0, stream>>>(cpad, BB * C2C * SZD * SZD);

  // one-time weight prep: f16 converts (layouts already fragment-native) + packed conv A
  k_prep_w2p<<<(C2C * KPACK + 255) / 256, blk, 0, stream>>>(conv2_w, w2p);
  k_prep_wd2p<<<(C1C * KPACK + 255) / 256, blk, 0, stream>>>(deconv2_w, wd2p);
  k_prep_wrc<<<(C1C * KW * KW + 255) / 256, blk, 0, stream>>>(recon_w, wrcf);
  k_f2h<<<(NREC * NCONV2 + 255) / 256, blk, 0, stream>>>(ff_in_w, wfih, NREC * NCONV2);
  k_f2h<<<(NREC * NREC + 255) / 256, blk, 0, stream>>>(ff_rec_w, wfrh, NREC * NREC);
  k_f2h<<<(NCONV2 * NREC + 255) / 256, blk, 0, stream>>>(ff_out_w, wfoh, NCONV2 * NREC);

  // 50-step recurrence
  for (int t = 0; t < TT; ++t) {
    const float* xt = xin + (size_t)t * BB * SZD * SZD;
    float* out_rec = out + (size_t)t * BB * NREC;
    float* out_rc  = out + (size_t)TT * BB * NREC + (size_t)t * BB * SZD * SZD;
    k_conv1_lif<<<(BB * C1C * SP1) / 256, blk, 0, stream>>>(xt, conv1_w, conv1_b, osc, t,
                                                            mem_c1, spk1);
    k_pack_conv2<<<(C1C * BB * SP2) / 256, blk, 0, stream>>>(spk1, bpack);
    k_conv_gemm_lif<SP2><<<(BB * SP2 / 16) / 8, blk, 0, stream>>>(w2p, bpack, conv2_b,
                                                                  osc, t, mem_c2, spk2);
    k_ff_gemm<<<16, blk, 0, stream>>>(spk2, wfih, srec[t & 1], wfrh, partial);
    k_ff_lif_reduce<<<(BB * NREC) / 256, blk, 0, stream>>>(partial, ff_in_b, ff_rec_b,
                                                           osc, t, mem_rec,
                                                           srec[(t + 1) & 1], out_rec);
    k_ffout_pad<<<(NCONV2 / 16) / 8, blk, 0, stream>>>(srec[(t + 1) & 1], wfoh, ff_out_b,
                                                       cpad);
    k_pack_deconv2<<<(C2C * BB * SP1) / 256, blk, 0, stream>>>(cpad, bpack);
    k_conv_gemm_lif<SP1><<<(BB * SP1 / 16) / 8, blk, 0, stream>>>(wd2p, bpack, deconv2_b,
                                                                  osc, t, mem_d2, spkd2);
    k_recon_lif<<<(BB * SZD * SZD) / 256, blk, 0, stream>>>(spkd2, wrcf, recon_b, osc, t,
                                                            mem_rc, out_rc);
  }
}
